// DecoderRNN_32349693673507
// MI455X (gfx1250) — matrix operand order repaired and checked by reading, not compiled, not run on hardware
//
#include <hip/hip_runtime.h>
#include <hip/hip_bf16.h>
#include <math.h>

typedef __attribute__((ext_vector_type(16))) _Float16 v16h;
typedef __attribute__((ext_vector_type(8)))  _Float16 h8;
typedef __attribute__((ext_vector_type(4)))  _Float16 h4;
typedef __attribute__((ext_vector_type(2)))  _Float16 h2;
typedef __attribute__((ext_vector_type(8)))  float    v8f;

// Problem constants
#define Bn   32
#define Pp   196
#define ENCd 2048
#define Tt   30
#define Vv   10000
#define Ee   512
#define Aa   512
#define Hh   512

#define AS_STRIDE 40   // halves per A-tile row (32 + pad); 80B rows -> 16B aligned
#define BS_STRIDE 40   // halves per Bt row; even -> k-pair stores 4B aligned

__device__ __forceinline__ float sigmoidf_(float x) { return 1.0f / (1.0f + __expf(-x)); }

// ---------------------------------------------------------------------------
// WMMA GEMM: C[M,N] = act( A[M,K](f32) @ B[K,N](f32) + bias + beta*C )
// M = gridDim.y*32, block = 256 (8 waves -> 2x4 grid of 16x16 WMMA tiles).
// f32 operands converted to f16 in double-buffered LDS; f32 accumulate via
// v_wmma_f32_16x16x32_f16. K % 32 == 0. N % 4 == 0 (edge via clamp+select).
// ---------------------------------------------------------------------------
__global__ void gemm32_wmma(const float* __restrict__ A, int lda,
                            const float* __restrict__ B, int ldb,
                            const float* __restrict__ bias,
                            float* __restrict__ C, long long ldc,
                            int N, int K, int beta, int act) {
    __shared__ __align__(16) _Float16 As[2][32 * AS_STRIDE];
    __shared__ __align__(16) _Float16 Bs[2][64 * BS_STRIDE];

    const int tid  = threadIdx.x;
    const int wave = tid >> 5;
    const int lane = tid & 31;
    const int mw   = wave >> 2;      // 0..1  (M subtile)
    const int nw   = wave & 3;       // 0..3  (N subtile)
    const int n0   = blockIdx.x * 64;
    const int m0   = blockIdx.y * 32;

    // --- A staging role: 1 row x 4 k per thread (coalesced float4) ---
    const int arow_st = tid >> 3;          // 0..31
    const int ak4     = (tid & 7) << 2;    // 0,4,..,28
    // --- B staging role: 2 k-rows x 4 n per thread (coalesced float4 per row) ---
    const int bn4 = (tid & 15) << 2;       // 0,4,..,60
    const int bkp = (tid >> 4) << 1;       // 0,2,..,30
    const int ncol   = n0 + bn4;
    const bool bvld  = (ncol + 3 < N);                 // group fully valid (N%4==0)
    int nclamp = ncol; if (nclamp > N - 4) nclamp = N - 4;

    // --- fragment-read addresses (constant across K chunks) ---
    const int arow  = mw * 16 + (lane & 15);
    const int koffA = (lane < 16) ? 0 : 8;             // A: K {0..7,16..23} / {8..15,24..31}
    const int bcol  = nw * 16 + (lane & 15);
    const int kb    = (lane < 16) ? 0 : 8;             // B: K {0..7,16..23} / {8..15,24..31}

    v8f acc = {0.f, 0.f, 0.f, 0.f, 0.f, 0.f, 0.f, 0.f};

    const int nchunks = K >> 5;

    // ---- preload chunk 0 into buffer 0 ----
    {
        const float4 av = *(const float4*)(A + (size_t)(m0 + arow_st) * lda + ak4);
        *(h4*)&As[0][arow_st * AS_STRIDE + ak4] =
            (h4){(_Float16)av.x, (_Float16)av.y, (_Float16)av.z, (_Float16)av.w};

        const float4 b0 = *(const float4*)(B + (size_t)bkp       * ldb + nclamp);
        const float4 b1 = *(const float4*)(B + (size_t)(bkp + 1) * ldb + nclamp);
        const float b0a[4] = {b0.x, b0.y, b0.z, b0.w};
        const float b1a[4] = {b1.x, b1.y, b1.z, b1.w};
        #pragma unroll
        for (int j = 0; j < 4; ++j) {
            const float u = bvld ? b0a[j] : 0.f;
            const float w = bvld ? b1a[j] : 0.f;
            *(h2*)&Bs[0][(bn4 + j) * BS_STRIDE + bkp] = (h2){(_Float16)u, (_Float16)w};
        }
    }

    int cur = 0;
    for (int kc = 0; kc < nchunks; ++kc) {
        __syncthreads();                       // buf[cur] writes visible

        // ---- issue next chunk's global loads early (overlap with WMMA) ----
        float4 av, b0, b1;
        const bool have_next = (kc + 1 < nchunks);
        if (have_next) {
            const int kn = (kc + 1) << 5;
            av = *(const float4*)(A + (size_t)(m0 + arow_st) * lda + kn + ak4);
            b0 = *(const float4*)(B + (size_t)(kn + bkp)     * ldb + nclamp);
            b1 = *(const float4*)(B + (size_t)(kn + bkp + 1) * ldb + nclamp);
        }

        // ---- fragments from buf[cur] + WMMA ----
        {
            union { v16h v; h8 h[2]; } af;
            af.h[0] = *(const h8*)(&As[cur][arow * AS_STRIDE + koffA]);
            af.h[1] = *(const h8*)(&As[cur][arow * AS_STRIDE + koffA + 16]);
            union { v16h v; h8 h[2]; } bf;
            bf.h[0] = *(const h8*)(&Bs[cur][bcol * BS_STRIDE + kb]);
            bf.h[1] = *(const h8*)(&Bs[cur][bcol * BS_STRIDE + kb + 16]);
            acc = __builtin_amdgcn_wmma_f32_16x16x32_f16(
                false, af.v, false, bf.v, (short)0, acc, false, false);
        }

        // ---- store next chunk into buf[cur^1] ----
        if (have_next) {
            const int nb = cur ^ 1;
            *(h4*)&As[nb][arow_st * AS_STRIDE + ak4] =
                (h4){(_Float16)av.x, (_Float16)av.y, (_Float16)av.z, (_Float16)av.w};
            const float b0a[4] = {b0.x, b0.y, b0.z, b0.w};
            const float b1a[4] = {b1.x, b1.y, b1.z, b1.w};
            #pragma unroll
            for (int j = 0; j < 4; ++j) {
                const float u = bvld ? b0a[j] : 0.f;
                const float w = bvld ? b1a[j] : 0.f;
                *(h2*)&Bs[nb][(bn4 + j) * BS_STRIDE + bkp] = (h2){(_Float16)u, (_Float16)w};
            }
        }
        cur ^= 1;
    }

    // ---- epilogue: C VGPR r -> (M = 16*mw + r + (lane>=16?8:0), N = 16*nw + lane%16)
    const int col  = n0 + nw * 16 + (lane & 15);
    const int rofs = mw * 16 + ((lane < 16) ? 0 : 8);
    if (col < N) {
        const float bv = bias ? bias[col] : 0.0f;
        #pragma unroll
        for (int r = 0; r < 8; ++r) {
            const int row = m0 + rofs + r;
            const long long off = (long long)row * ldc + col;
            float v = acc[r] + bv;
            if (beta) v += C[off];
            if (act == 1) v = sigmoidf_(v);
            C[off] = v;
        }
    }
}

// ---------------------------------------------------------------------------
// mean over P positions: mean[b,d] = (1/196) * sum_p enc[b,p,d]
// ---------------------------------------------------------------------------
__global__ void mean_kernel(const float* __restrict__ enc, float* __restrict__ meanb) {
    const int b = blockIdx.x;
    const int d = blockIdx.y * 256 + threadIdx.x;
    const float* e = enc + (size_t)b * Pp * ENCd + d;
    float acc = 0.f;
    for (int p = 0; p < Pp; ++p) acc += e[(size_t)p * ENCd];
    meanb[(size_t)b * ENCd + d] = acc * (1.0f / (float)Pp);
}

// ---------------------------------------------------------------------------
// attention scores + softmax; one block per batch element.
// ---------------------------------------------------------------------------
__global__ void attention_kernel(const float* __restrict__ att1,
                                 const float* __restrict__ att2,
                                 const float* __restrict__ W_full,
                                 const float* __restrict__ b_full,
                                 float* __restrict__ alpha,
                                 float* __restrict__ alphas_out, int t) {
    const int b = blockIdx.x;
    __shared__ float att2_s[Aa];
    __shared__ float wf_s[Aa];
    __shared__ float e_s[Pp];
    __shared__ float red[256];
    const int tid = threadIdx.x;

    for (int j = tid; j < Aa; j += 256) {
        att2_s[j] = att2[(size_t)b * Aa + j];
        wf_s[j]   = W_full[j];
    }
    __syncthreads();

    const int wave = tid >> 5, lane = tid & 31;
    for (int p = wave; p < Pp; p += 8) {
        const float* row = att1 + ((size_t)b * Pp + p) * Aa;
        float acc = 0.f;
        for (int k = lane; k < Aa; k += 32) {
            float v = row[k] + att2_s[k];
            v = v > 0.f ? v : 0.f;
            acc += v * wf_s[k];
        }
        #pragma unroll
        for (int o = 16; o > 0; o >>= 1) acc += __shfl_xor(acc, o, 32);
        if (lane == 0) e_s[p] = acc + b_full[0];
    }
    __syncthreads();

    red[tid] = (tid < Pp) ? e_s[tid] : -1e30f;
    __syncthreads();
    for (int s = 128; s > 0; s >>= 1) {
        if (tid < s) red[tid] = fmaxf(red[tid], red[tid + s]);
        __syncthreads();
    }
    const float mx = red[0];
    __syncthreads();
    const float ex = (tid < Pp) ? __expf(e_s[tid] - mx) : 0.f;
    red[tid] = ex;
    __syncthreads();
    for (int s = 128; s > 0; s >>= 1) {
        if (tid < s) red[tid] += red[tid + s];
        __syncthreads();
    }
    const float inv = 1.0f / red[0];
    if (tid < Pp) {
        const float a = ex * inv;
        alpha[(size_t)b * Pp + tid] = a;
        alphas_out[((size_t)b * Tt + t) * Pp + tid] = a;
    }
}

// ---------------------------------------------------------------------------
// awe[b,d] = sum_p alpha[b,p]*enc[b,p,d]; x[b, 512+d] = gate[b,d]*awe[b,d]
// ---------------------------------------------------------------------------
__global__ void awe_x_kernel(const float* __restrict__ alpha,
                             const float* __restrict__ enc,
                             const float* __restrict__ gate,
                             float* __restrict__ x) {
    const int b = blockIdx.x;
    const int d = blockIdx.y * 256 + threadIdx.x;
    __shared__ float al[Pp];
    if (threadIdx.x < Pp) al[threadIdx.x] = alpha[(size_t)b * Pp + threadIdx.x];
    __syncthreads();
    const float* e = enc + (size_t)b * Pp * ENCd + d;
    float acc = 0.f;
    for (int p = 0; p < Pp; ++p) acc += al[p] * e[(size_t)p * ENCd];
    x[(size_t)b * (Ee + ENCd) + Ee + d] = gate[(size_t)b * ENCd + d] * acc;
}

// ---------------------------------------------------------------------------
// x[b, 0:512] = emb[captions[b,t], :]
// ---------------------------------------------------------------------------
__global__ void emb_gather_kernel(const float* __restrict__ emb,
                                  const int* __restrict__ captions,
                                  float* __restrict__ x, int t) {
    const int b = blockIdx.x;
    const int cap = captions[b * Tt + t];
    for (int j = threadIdx.x; j < Ee; j += 256)
        x[(size_t)b * (Ee + ENCd) + j] = emb[(size_t)cap * Ee + j];
}

// ---------------------------------------------------------------------------
// LSTM pointwise update (torch gate order i,f,g,o), in-place h/c.
// ---------------------------------------------------------------------------
__global__ void lstm_kernel(const float* __restrict__ gates,
                            float* __restrict__ h, float* __restrict__ c) {
    const int idx = blockIdx.x * 256 + threadIdx.x;   // 0 .. 32*512-1
    const int b = idx >> 9, j = idx & 511;
    const float* g = gates + (size_t)b * (4 * Hh);
    const float ig = sigmoidf_(g[j]);
    const float fg = sigmoidf_(g[Hh + j]);
    const float gg = tanhf(g[2 * Hh + j]);
    const float og = sigmoidf_(g[3 * Hh + j]);
    const float cn = fg * c[idx] + ig * gg;
    c[idx] = cn;
    h[idx] = og * tanhf(cn);
}

// ---------------------------------------------------------------------------
extern "C" void kernel_launch(void* const* d_in, const int* in_sizes, int n_in,
                              void* d_out, int out_size, void* d_ws, size_t ws_size,
                              hipStream_t stream) {
    const float* enc       = (const float*)d_in[0];
    const int*   captions  = (const int*)  d_in[1];
    const float* emb       = (const float*)d_in[2];
    const float* W_enc_att = (const float*)d_in[3];
    const float* b_enc_att = (const float*)d_in[4];
    const float* W_dec_att = (const float*)d_in[5];
    const float* b_dec_att = (const float*)d_in[6];
    const float* W_full    = (const float*)d_in[7];
    const float* b_full    = (const float*)d_in[8];
    const float* W_ih      = (const float*)d_in[9];
    const float* b_ih      = (const float*)d_in[10];
    const float* W_hh      = (const float*)d_in[11];
    const float* b_hh      = (const float*)d_in[12];
    const float* W_init_h  = (const float*)d_in[13];
    const float* b_init_h  = (const float*)d_in[14];
    const float* W_init_c  = (const float*)d_in[15];
    const float* b_init_c  = (const float*)d_in[16];
    const float* W_fbeta   = (const float*)d_in[17];
    const float* b_fbeta   = (const float*)d_in[18];
    const float* W_fc      = (const float*)d_in[19];
    const float* b_fc      = (const float*)d_in[20];

    float* out = (float*)d_out;
    const long long PRED = (long long)Bn * Tt * Vv;          // 9,600,000
    float* alphas_out = out + PRED;

    // workspace layout (floats)
    float* ws    = (float*)d_ws;
    float* att1  = ws;                                        // 6272*512
    float* meanb = att1  + (size_t)Bn * Pp * Aa;              // 32*2048
    float* h     = meanb + (size_t)Bn * ENCd;                 // 32*512
    float* c     = h     + (size_t)Bn * Hh;                   // 32*512
    float* att2  = c     + (size_t)Bn * Hh;                   // 32*512
    float* alpha = att2  + (size_t)Bn * Aa;                   // 32*196
    float* gate  = alpha + (size_t)Bn * Pp;                   // 32*2048
    float* x     = gate  + (size_t)Bn * ENCd;                 // 32*2560
    float* gates = x     + (size_t)Bn * (Ee + ENCd);          // 32*2048

    const dim3 blk(256);

    // ---- setup ----
    mean_kernel<<<dim3(Bn, ENCd / 256), blk, 0, stream>>>(enc, meanb);
    gemm32_wmma<<<dim3(Hh / 64, 1), blk, 0, stream>>>(meanb, ENCd, W_init_h, Hh, b_init_h,
                                                      h, Hh, Hh, ENCd, 0, 0);
    gemm32_wmma<<<dim3(Hh / 64, 1), blk, 0, stream>>>(meanb, ENCd, W_init_c, Hh, b_init_c,
                                                      c, Hh, Hh, ENCd, 0, 0);
    gemm32_wmma<<<dim3(Aa / 64, (Bn * Pp) / 32), blk, 0, stream>>>(
        enc, ENCd, W_enc_att, Aa, b_enc_att, att1, Aa, Aa, ENCd, 0, 0);

    // ---- time steps ----
    for (int t = 0; t < Tt; ++t) {
        gemm32_wmma<<<dim3(Aa / 64, 1), blk, 0, stream>>>(h, Hh, W_dec_att, Aa, b_dec_att,
                                                          att2, Aa, Aa, Hh, 0, 0);
        attention_kernel<<<dim3(Bn), blk, 0, stream>>>(att1, att2, W_full, b_full,
                                                       alpha, alphas_out, t);
        gemm32_wmma<<<dim3(ENCd / 64, 1), blk, 0, stream>>>(h, Hh, W_fbeta, ENCd, b_fbeta,
                                                            gate, ENCd, ENCd, Hh, 0, 1);
        emb_gather_kernel<<<dim3(Bn), blk, 0, stream>>>(emb, captions, x, t);
        awe_x_kernel<<<dim3(Bn, ENCd / 256), blk, 0, stream>>>(alpha, enc, gate, x);
        gemm32_wmma<<<dim3((4 * Hh) / 64, 1), blk, 0, stream>>>(x, Ee + ENCd, W_ih, 4 * Hh,
                                                                b_ih, gates, 4 * Hh,
                                                                4 * Hh, Ee + ENCd, 0, 0);
        gemm32_wmma<<<dim3((4 * Hh) / 64, 1), blk, 0, stream>>>(h, Hh, W_hh, 4 * Hh,
                                                                b_hh, gates, 4 * Hh,
                                                                4 * Hh, Hh, 1, 0);
        lstm_kernel<<<dim3((Bn * Hh) / 256), blk, 0, stream>>>(gates, h, c);
        gemm32_wmma<<<dim3((Vv + 63) / 64, 1), blk, 0, stream>>>(
            h, Hh, W_fc, Vv, b_fc, out + (long long)t * Vv, (long long)Tt * Vv,
            Vv, Hh, 0, 0);
    }
}
